// Cvxnnregression_79671643341527
// MI455X (gfx1250) — matrix operand order repaired and checked by reading, not compiled, not run on hardware
//
#include <hip/hip_runtime.h>
#include <hip/hip_bf16.h>

typedef __attribute__((ext_vector_type(16))) _Float16 v16h;
typedef __attribute__((ext_vector_type(8)))  _Float16 v8h;
typedef __attribute__((ext_vector_type(2)))  _Float16 v2h;
typedef __attribute__((ext_vector_type(8)))  float    v8f;

#define WAVES_PER_BLOCK 2
#define THREADS (WAVES_PER_BLOCK * 32)

// Workspace layout (units = halves): pre-swizzled B-fragment tiles.
// Each tile = 32x16 (KxN) = 512 halves, lane-major: [lane][e0..e15].
// Hidden width padded 100 -> 112 (7 n-tiles of 16). K padded 100 -> 128.
//   W1: 1 k-block x 7 n-tiles  @ 0      (tile base 0)
//   W2: 4 k-blocks x 7 n-tiles @ 3584   (tile base 7)
//   W3: 4 k-blocks x 7 n-tiles @ 17920  (tile base 35)
//   Wr: 4 k-blocks x 1 n-tile  @ 32256  (tile base 63)
// Biases (f32, 128-padded) at byte offset 68608: b1@0, b2@128, b3@256, br@384.
#define OFF_W1 0
#define OFF_W2 3584
#define OFF_W3 17920
#define OFF_WR 32256
#define TOT_WH 34304
#define BIAS_BYTE_OFF 68608
#define TB_W1 0
#define TB_W2 7
#define TB_W3 35
#define TB_WR 63

__global__ void prep_kernel(const float* __restrict__ W1, const float* __restrict__ b1,
                            const float* __restrict__ W2, const float* __restrict__ b2,
                            const float* __restrict__ W3, const float* __restrict__ b3,
                            const float* __restrict__ Wr, const float* __restrict__ br,
                            _Float16* __restrict__ wsw, float* __restrict__ wsb)
{
    int idx = blockIdx.x * blockDim.x + threadIdx.x;
    if (idx < TOT_WH) {
        const float* W; int off, Kr, Nr, ntiles;
        if (idx < OFF_W2)      { W = W1; off = OFF_W1; Kr = 16;  Nr = 100; ntiles = 7; }
        else if (idx < OFF_W3) { W = W2; off = OFF_W2; Kr = 100; Nr = 100; ntiles = 7; }
        else if (idx < OFF_WR) { W = W3; off = OFF_W3; Kr = 100; Nr = 100; ntiles = 7; }
        else                   { W = Wr; off = OFF_WR; Kr = 100; Nr = 4;   ntiles = 1; }
        int local = idx - off;
        int t  = local >> 9;       // tile index within region
        int r  = local & 511;
        int ln = r >> 4;           // lane 0..31
        int e  = r & 15;           // element 0..15
        int kt = t / ntiles, nt = t % ntiles;
        // B layout: lane -> N = ln&15 ; element e -> K = (e<8 ? 8*(ln>>4)+e : 16+8*(ln>>4)+(e-8))
        int k = kt * 32 + (e < 8 ? (ln >> 4) * 8 + e : 16 + (ln >> 4) * 8 + (e - 8));
        int n = nt * 16 + (ln & 15);
        float v = (k < Kr && n < Nr) ? W[k * Nr + n] : 0.0f;
        wsw[idx] = (_Float16)v;
    } else if (idx < TOT_WH + 400) {
        int j = idx - TOT_WH;
        float v = 0.0f;
        if (j < 128)      { if (j < 100)      v = b1[j]; }
        else if (j < 256) { int n = j - 128; if (n < 100) v = b2[n]; }
        else if (j < 384) { int n = j - 256; if (n < 100) v = b3[n]; }
        else              { int n = j - 384; if (n < 4)   v = br[n]; }
        wsb[j] = v;
    }
}

// Packed bias-add + ReLU + f16 convert for an accumulator element pair (rows
// v, v+1 of the same column): v_cvt_pk_f16_f32 + v_pk_add_f16 + v_pk_max_num_f16.
__device__ __forceinline__ void store_pair_relu(_Float16* __restrict__ dst, int baseRow,
                                                int col, float lo, float hi, v2h b2)
{
    v2h p = {(_Float16)lo, (_Float16)hi};
    p = p + b2;
    v2h z2 = {};
    p = __builtin_elementwise_max(p, z2);
    dst[baseRow * 128 + col]       = p[0];
    dst[(baseRow + 1) * 128 + col] = p[1];
}

__launch_bounds__(THREADS)
__global__ void mlp_head_kernel(const float* __restrict__ x,
                                const _Float16* __restrict__ wsw,
                                const float* __restrict__ wsb,
                                float* __restrict__ out, int Btot)
{
    // [wave][buffer][row-tile][16 rows x 128 cols]
    __shared__ __align__(16) _Float16 lds_h[WAVES_PER_BLOCK][2][2][16 * 128];
    __shared__ __align__(16) float    lds_r[WAVES_PER_BLOCK][2][16 * 16];

    const int lane = threadIdx.x & 31;
    const int wave = threadIdx.x >> 5;
    const int pair = blockIdx.x * WAVES_PER_BLOCK + wave;  // 32-sample tile pair
    const int m0   = pair * 32;
    if (m0 >= Btot) return;
    const int m = lane & 15;   // row for A-frags / col (N) for B-, C-frags
    const int g = lane >> 4;   // half-wave group
    const int n = m;

    const v16h* wt = (const v16h*)wsw;   // tile t, lane l -> wt[t*32 + l]

    // ---- zero-init pad columns 112..127 of both buffers / both row-tiles.
    // Weights are zero for k>=100, so pad only needs to avoid NaN*0.
    {
        v8h z = {};
        _Float16* a0 = &lds_h[wave][0][g][0];
        _Float16* a1 = &lds_h[wave][1][g][0];
        *(v8h*)(a0 + m * 128 + 112) = z;
        *(v8h*)(a0 + m * 128 + 120) = z;
        *(v8h*)(a1 + m * 128 + 112) = z;
        *(v8h*)(a1 + m * 128 + 120) = z;
    }

    // ---- A fragments of x for both row-tiles: 16x32 f16 (K>=16 zero pad)
    // A layout: element e -> K = (e<8 ? 8g+e : 16+8g+(e-8))
    v16h ax[2] = {{}, {}};
#pragma unroll
    for (int t = 0; t < 2; ++t) {
        const float4* xr = (const float4*)(x + (size_t)(m0 + t * 16 + m) * 16);
        float4 q0 = xr[g * 2 + 0];
        float4 q1 = xr[g * 2 + 1];
        float f[8] = {q0.x, q0.y, q0.z, q0.w, q1.x, q1.y, q1.z, q1.w};
#pragma unroll
        for (int i = 0; i < 8; ++i) ax[t][i] = (_Float16)f[i];
    }

    // ---- layer 1: h1 = relu(x @ W1 + b1) -> buffer 0, cols 0..111
#pragma unroll
    for (int nt = 0; nt < 7; ++nt) {
        float bv = wsb[nt * 16 + n];
        _Float16 bh = (_Float16)bv;
        v2h b2 = {bh, bh};
        v8f a0 = {};   // C = inline 0; bias added in packed-f16 epilogue
        v8f a1 = {};
        v16h bf = wt[(TB_W1 + nt) * 32 + lane];
        a0 = __builtin_amdgcn_wmma_f32_16x16x32_f16(false, ax[0], false, bf,
                                                    (short)0, a0, false, false);
        a1 = __builtin_amdgcn_wmma_f32_16x16x32_f16(false, ax[1], false, bf,
                                                    (short)0, a1, false, false);
        _Float16* h0 = &lds_h[wave][0][0][0];
        _Float16* h1 = &lds_h[wave][0][1][0];
        const int col = nt * 16 + n;
#pragma unroll
        for (int v = 0; v < 8; v += 2) {   // C layout: VGPR v -> row M = v+8g
            store_pair_relu(h0, v + 8 * g, col, a0[v], a0[v + 1], b2);
            store_pair_relu(h1, v + 8 * g, col, a1[v], a1[v + 1], b2);
        }
    }

    // ---- layers 2 and 3: h = relu(h @ W + b), K = 128 (4 k-blocks)
#pragma unroll
    for (int layer = 0; layer < 2; ++layer) {
        const int sb = layer;          // source buffer: 0 then 1
        const int db = layer ^ 1;      // dest buffer:   1 then 0
        const int wbase = (layer == 0) ? TB_W2 : TB_W3;
        const int bbase = (layer == 0) ? 128 : 256;
        v16h af[2][4];
#pragma unroll
        for (int t = 0; t < 2; ++t)
#pragma unroll
            for (int kt = 0; kt < 4; ++kt) {
                const _Float16* src = &lds_h[wave][sb][t][0];
                v8h lo = *(const v8h*)(src + m * 128 + kt * 32 + g * 8);
                v8h hi = *(const v8h*)(src + m * 128 + kt * 32 + 16 + g * 8);
                af[t][kt] = __builtin_shufflevector(lo, hi, 0, 1, 2, 3, 4, 5, 6, 7,
                                                    8, 9, 10, 11, 12, 13, 14, 15);
            }
#pragma unroll
        for (int nt = 0; nt < 7; ++nt) {
            float bv = wsb[bbase + nt * 16 + n];
            _Float16 bh = (_Float16)bv;
            v2h b2 = {bh, bh};
            v8f a0 = {};
            v8f a1 = {};
#pragma unroll
            for (int kt = 0; kt < 4; ++kt) {
                v16h bf = wt[(wbase + kt * 7 + nt) * 32 + lane];
                a0 = __builtin_amdgcn_wmma_f32_16x16x32_f16(false, af[0][kt], false, bf,
                                                            (short)0, a0, false, false);
                a1 = __builtin_amdgcn_wmma_f32_16x16x32_f16(false, af[1][kt], false, bf,
                                                            (short)0, a1, false, false);
            }
            _Float16* h0 = &lds_h[wave][db][0][0];
            _Float16* h1 = &lds_h[wave][db][1][0];
            const int col = nt * 16 + n;
#pragma unroll
            for (int v = 0; v < 8; v += 2) {
                store_pair_relu(h0, v + 8 * g, col, a0[v], a0[v + 1], b2);
                store_pair_relu(h1, v + 8 * g, col, a1[v], a1[v + 1], b2);
            }
        }
    }

    // ---- output layer: r = h3 @ Wr + br (h3 in buffer 0; cols 0..3 real).
    // Keep exact f32 bias in C here (r_predict is an output).
    {
        v16h af[2][4];
#pragma unroll
        for (int t = 0; t < 2; ++t)
#pragma unroll
            for (int kt = 0; kt < 4; ++kt) {
                const _Float16* src = &lds_h[wave][0][t][0];
                v8h lo = *(const v8h*)(src + m * 128 + kt * 32 + g * 8);
                v8h hi = *(const v8h*)(src + m * 128 + kt * 32 + 16 + g * 8);
                af[t][kt] = __builtin_shufflevector(lo, hi, 0, 1, 2, 3, 4, 5, 6, 7,
                                                    8, 9, 10, 11, 12, 13, 14, 15);
            }
        float bv = wsb[384 + n];
        v8f a0 = {bv, bv, bv, bv, bv, bv, bv, bv};
        v8f a1 = a0;
#pragma unroll
        for (int kt = 0; kt < 4; ++kt) {
            v16h bf = wt[(TB_WR + kt) * 32 + lane];
            a0 = __builtin_amdgcn_wmma_f32_16x16x32_f16(false, af[0][kt], false, bf,
                                                        (short)0, a0, false, false);
            a1 = __builtin_amdgcn_wmma_f32_16x16x32_f16(false, af[1][kt], false, bf,
                                                        (short)0, a1, false, false);
        }
#pragma unroll
        for (int v = 0; v < 8; ++v) {
            lds_r[wave][0][(v + 8 * g) * 16 + n] = a0[v];
            lds_r[wave][1][(v + 8 * g) * 16 + n] = a1[v];
        }
    }

    // ---- head: one sample per lane (all 32 lanes). LDS ops in same wave are
    // in-order, so the r-tile stores above are visible without a barrier.
    {
        const int s   = m0 + lane;
        const int rt  = lane >> 4;     // which r-tile
        const int row = lane & 15;
        const float* x0 = x + (size_t)s * 16;
        const float r0 = lds_r[wave][rt][row * 16 + 0];
        const float r1 = lds_r[wave][rt][row * 16 + 1];
        const float r2 = lds_r[wave][rt][row * 16 + 2];
        const float r3 = lds_r[wave][rt][row * 16 + 3];
        const float rb0 = x0[12], rb1 = x0[13];
        const float den0 = r0 + r1, den1 = r2 + r3;
        const float rt0 = rb0 * r0 / den0;
        const float rt1 = rb0 * r1 / den0;
        const float rt2 = rb1 * r2 / den1;
        const float rt3 = rb1 * r3 / den1;
        float p0, p1, p2, p3;
        {   // pair 1: G = x0[0:4], sigma = x0[8:10], rp = (rt0, rt1)
            float g00 = x0[0], g01 = x0[1], g10 = x0[2], g11 = x0[3];
            float s0 = x0[8], s1 = x0[9];
            float d0i = (g00 != 0.0f) ? 1.0f / g00 : 0.0f;
            float d1i = (g11 != 0.0f) ? 1.0f / g11 : 0.0f;
            float F01 = d0i * g01, F10 = d1i * g10;
            float v0 = d0i * s0, v1 = d1i * s1;
            float rhs0 = rt0 * v0, rhs1 = rt1 * v1;
            float det = 1.0f - (rt0 * F01) * (rt1 * F10);
            p0 = (rhs0 + rt0 * F01 * rhs1) / det;
            p1 = (rt1 * F10 * rhs0 + rhs1) / det;
        }
        {   // pair 2: G = x0[4:8], sigma = x0[10:12], rp = (rt2, rt3)
            float g00 = x0[4], g01 = x0[5], g10 = x0[6], g11 = x0[7];
            float s0 = x0[10], s1 = x0[11];
            float d0i = (g00 != 0.0f) ? 1.0f / g00 : 0.0f;
            float d1i = (g11 != 0.0f) ? 1.0f / g11 : 0.0f;
            float F01 = d0i * g01, F10 = d1i * g10;
            float v0 = d0i * s0, v1 = d1i * s1;
            float rhs0 = rt2 * v0, rhs1 = rt3 * v1;
            float det = 1.0f - (rt2 * F01) * (rt3 * F10);
            p2 = (rhs0 + rt2 * F01 * rhs1) / det;
            p3 = (rt3 * F10 * rhs0 + rhs1) / det;
        }
        float* po = out + (size_t)s * 4;
        po[0] = fmaxf(p0, 0.0f);
        po[1] = fmaxf(p1, 0.0f);
        po[2] = fmaxf(p2, 0.0f);
        po[3] = fmaxf(p3, 0.0f);
        float* ro = out + (size_t)Btot * 4 + (size_t)s * 4;
        ro[0] = r0; ro[1] = r1; ro[2] = r2; ro[3] = r3;
    }
}

extern "C" void kernel_launch(void* const* d_in, const int* in_sizes, int n_in,
                              void* d_out, int out_size, void* d_ws, size_t ws_size,
                              hipStream_t stream)
{
    const float* x  = (const float*)d_in[0];
    const float* W1 = (const float*)d_in[1];
    const float* b1 = (const float*)d_in[2];
    const float* W2 = (const float*)d_in[3];
    const float* b2 = (const float*)d_in[4];
    const float* W3 = (const float*)d_in[5];
    const float* b3 = (const float*)d_in[6];
    const float* Wr = (const float*)d_in[7];
    const float* br = (const float*)d_in[8];
    float* out = (float*)d_out;
    const int B = in_sizes[0] / 16;

    _Float16* wsw = (_Float16*)d_ws;
    float*    wsb = (float*)((char*)d_ws + BIAS_BYTE_OFF);

    const int prepThreads = TOT_WH + 400;
    prep_kernel<<<(prepThreads + 255) / 256, 256, 0, stream>>>(
        W1, b1, W2, b2, W3, b3, Wr, br, wsw, wsb);

    const int pairs  = (B + 31) / 32;
    const int blocks = (pairs + WAVES_PER_BLOCK - 1) / WAVES_PER_BLOCK;
    mlp_head_kernel<<<blocks, THREADS, 0, stream>>>(x, wsw, wsb, out, B);
}
